// CannyLoss_60318520705682
// MI455X (gfx1250) — compile-verified, matrix operand order checked by reading, not run on hardware
//
#include <hip/hip_runtime.h>
#include <cstdint>

// ---------------------------------------------------------------------------
// CannyLoss for MI455X (gfx1250, wave32).
//
// Pipeline (all bit-packed: bit i of word w == pixel x = 32*w + i):
//   K1 edge_kernel : binarize -> 3x3 Sobel (replicate border) -> L1 mag ->
//                    NMS (zero-border neighbor mags) -> weak/strong bits,
//                    packed per-wave via ballot_w32 (wave32 => 1 word/wave).
//   K2 hyst_kernel : one workgroup per (image, pred/label). strong+weak
//                    bitplanes live in LDS (2 x 32KB << 320KB WGP LDS),
//                    loaded via gfx1250 async global->LDS, 16 two-phase
//                    dilate8 iterations (matches reference), async store-back.
//   K3 count_kernel: popcount(strong_pred ^ strong_label); wave reduction
//                    through v_wmma_f32_16x16x32_f16 (A = lane count in one
//                    K slot, B = ones); exact float atomicAdd (integers).
//   K4 finalize    : loss = count * (-log(1e-38)) / (32*512*512).
//
// Workspace layout (uint32): strong[64*8192] | weak[64*8192] | counter(f32)
//   = 4 MB + 4 B. Plane q = which*32 + b, word = q*8192 + y*16 + (x>>5).
// ---------------------------------------------------------------------------

#define IMG_B 32
#define IMG_H 512
#define IMG_W 512
#define WPR   16            // 512 bits / 32 = 16 words per row
#define PLANE_WORDS (IMG_H * WPR)          // 8192 words per image plane
#define ALL_PLANES  (64 * PLANE_WORDS)     // pred(32) + labels(32)
#define HALF_PLANES (32 * PLANE_WORDS)     // offset from pred to labels
#define NPIX (IMG_B * IMG_H * IMG_W)       // 8388608

typedef __attribute__((ext_vector_type(16))) _Float16 v16h;
typedef __attribute__((ext_vector_type(8)))  float    v8f;

// ---- gfx1250 async global<->LDS path (guarded; plain-load fallback) -------
#if defined(__HIP_DEVICE_COMPILE__) && defined(__gfx1250__)
#if __has_builtin(__builtin_amdgcn_global_load_async_to_lds_b32) && \
    __has_builtin(__builtin_amdgcn_global_store_async_from_lds_b32)
#define USE_ASYNC_LDS 1
#endif
#endif

#if defined(USE_ASYNC_LDS)
// Probe-confirmed parameter types: (AS1 int* global, AS3 int* lds, imm, imm)
typedef __attribute__((address_space(1))) int as1_int_t;
typedef __attribute__((address_space(3))) int as3_int_t;
__device__ __forceinline__ void async_g2l(uint32_t* lds, const uint32_t* g) {
    __builtin_amdgcn_global_load_async_to_lds_b32(
        (as1_int_t*)g, (as3_int_t*)lds, 0, 0);
}
__device__ __forceinline__ void async_l2g(uint32_t* g, const uint32_t* lds) {
    __builtin_amdgcn_global_store_async_from_lds_b32(
        (as1_int_t*)g, (as3_int_t*)lds, 0, 0);
}
__device__ __forceinline__ void wait_asynccnt0() {
#if __has_builtin(__builtin_amdgcn_s_wait_asynccnt)
    __builtin_amdgcn_s_wait_asynccnt(0);
#else
    asm volatile("s_wait_asynccnt 0" ::: "memory");
#endif
}
#endif

__device__ __forceinline__ unsigned ballot32(bool p) {
#if defined(__HIP_DEVICE_COMPILE__) && __has_builtin(__builtin_amdgcn_ballot_w32)
    return __builtin_amdgcn_ballot_w32(p);
#else
    return (unsigned)__ballot(p);
#endif
}

// Wave32 sum reduction through the matrix pipe: D = A x ones + 0.
// Lane m (m<16) holds A[m][K0]=v, lane m+16 holds A[m][K8]=v (16-bit A
// layout, ISA 7.12.2). Row-m sums appear replicated across N; per-lane sum
// of the 8 D VGPRs + lanes {0,16} covers all 32 lanes exactly once.
__device__ __forceinline__ float wave_reduce_wmma(float v) {
    v16h a = {};
    a[0] = (_Float16)v;           // exact: v is an integer <= 2048
    v16h ones;
#pragma unroll
    for (int i = 0; i < 16; ++i) ones[i] = (_Float16)1.0f;
    v8f c = {};
    c = __builtin_amdgcn_wmma_f32_16x16x32_f16(
        /*neg_a=*/false, a, /*neg_b=*/false, ones,
        /*c_mod=*/(short)0, c, /*reuse_a=*/false, /*reuse_b=*/false);
    float s = c[0] + c[1] + c[2] + c[3] + c[4] + c[5] + c[6] + c[7];
    return __shfl(s, 0, 32) + __shfl(s, 16, 32);
}

__device__ __forceinline__ int clampi(int v, int lo, int hi) {
    return v < lo ? lo : (v > hi ? hi : v);
}

// ---------------- K1: binarize + Sobel + NMS -> packed weak/strong --------
__global__ __launch_bounds__(256) void edge_kernel(
    const float* __restrict__ pred, const float* __restrict__ labels,
    uint32_t* __restrict__ S, uint32_t* __restrict__ Wk,
    float* __restrict__ counter)
{
    if (blockIdx.x == 0 && blockIdx.y == 0 && blockIdx.z == 0 && threadIdx.x == 0)
        *counter = 0.0f;   // zeroed before K3 runs (stream order)

    const int which = (int)blockIdx.z >> 5;          // 0 = pred, 1 = labels
    const int b     = (int)blockIdx.z & 31;
    const int x     = (int)blockIdx.x * 256 + (int)threadIdx.x;
    const int y     = (int)blockIdx.y;
    const float* base = (which ? labels : pred) + (size_t)b * (IMG_H * IMG_W);

    // 5x5 binarized patch, replicate border (exactly floor(v/0.5000001)*255)
    float p[5][5];
#pragma unroll
    for (int i = 0; i < 5; ++i) {
        const float* row = base + clampi(y - 2 + i, 0, IMG_H - 1) * IMG_W;
#pragma unroll
        for (int j = 0; j < 5; ++j) {
            float v = row[clampi(x - 2 + j, 0, IMG_W - 1)];
            p[i][j] = floorf(v / 0.5000001f) * 255.0f;
        }
    }

    // 3x3 of L1-gradient magnitudes (zero outside image, for NMS neighbors)
    float gx0 = 0.f, gy0 = 0.f, m[3][3];
#pragma unroll
    for (int dy = -1; dy <= 1; ++dy) {
#pragma unroll
        for (int dx = -1; dx <= 1; ++dx) {
            const int ci = 2 + dy, cj = 2 + dx;
            float gx = (p[ci-1][cj+1] + 2.f*p[ci][cj+1] + p[ci+1][cj+1])
                     - (p[ci-1][cj-1] + 2.f*p[ci][cj-1] + p[ci+1][cj-1]);
            float gy = (p[ci+1][cj-1] + 2.f*p[ci+1][cj] + p[ci+1][cj+1])
                     - (p[ci-1][cj-1] + 2.f*p[ci-1][cj] + p[ci-1][cj+1]);
            float mag = fabsf(gx) + fabsf(gy);
            bool inb = ((unsigned)(y + dy) < (unsigned)IMG_H) &&
                       ((unsigned)(x + dx) < (unsigned)IMG_W);
            m[dy+1][dx+1] = inb ? mag : 0.f;
            if (dy == 0 && dx == 0) { gx0 = gx; gy0 = gy; }
        }
    }

    const float mag0 = m[1][1];
    const float ax = fabsf(gx0), ay = fabsf(gy0);
    const bool is_h     = ay <= 0.4142135623730951f * ax;
    const bool is_v     = ay >= 2.414213562373095f  * ax;
    const bool diag_pos = (gx0 * gy0) >= 0.0f;
    const float n1 = is_h ? m[1][0] : (is_v ? m[0][1] : (diag_pos ? m[0][0] : m[0][2]));
    const float n2 = is_h ? m[1][2] : (is_v ? m[2][1] : (diag_pos ? m[2][2] : m[2][0]));
    const bool keep   = (mag0 > n1) && (mag0 >= n2);
    const bool weak   = keep && (mag0 > 30.0f);
    const bool strong = keep && (mag0 > 150.0f);

    const unsigned sm = ballot32(strong);
    const unsigned wm = ballot32(weak);
    if ((threadIdx.x & 31u) == 0u) {
        const int q    = which * 32 + b;
        const int word = ((q * IMG_H + y) << 4) + (x >> 5);
        S[word]  = sm;
        Wk[word] = wm;
    }
}

// ---------------- K2: hysteresis, whole bitplane resident in LDS ----------
__global__ __launch_bounds__(1024) void hyst_kernel(
    uint32_t* __restrict__ S, const uint32_t* __restrict__ Wk)
{
    __shared__ uint32_t ls[PLANE_WORDS];   // 32 KB
    __shared__ uint32_t lw[PLANE_WORDS];   // 32 KB
    const int q   = (int)blockIdx.x;
    uint32_t* gS       = S  + (size_t)q * PLANE_WORDS;
    const uint32_t* gW = Wk + (size_t)q * PLANE_WORDS;
    const int tid = (int)threadIdx.x;

#pragma unroll
    for (int k = 0; k < 8; ++k) {
        const int idx = tid + k * 1024;
#if defined(USE_ASYNC_LDS)
        async_g2l(&ls[idx], &gS[idx]);
        async_g2l(&lw[idx], &gW[idx]);
#else
        ls[idx] = gS[idx];
        lw[idx] = gW[idx];
#endif
    }
#if defined(USE_ASYNC_LDS)
    wait_asynccnt0();
#endif
    __syncthreads();

    for (int it = 0; it < 16; ++it) {
        uint32_t nv[8];
#pragma unroll
        for (int k = 0; k < 8; ++k) {
            const int idx = tid + k * 1024;
            const int y = idx >> 4, c = idx & 15;
            uint32_t d = 0u;
#pragma unroll
            for (int r = -1; r <= 1; ++r) {
                const int ry = y + r;
                if ((unsigned)ry < (unsigned)IMG_H) {
                    const int bi = (ry << 4) + c;
                    const uint32_t mid = ls[bi];
                    const uint32_t lft = (c > 0)  ? ls[bi - 1] : 0u;
                    const uint32_t rgt = (c < 15) ? ls[bi + 1] : 0u;
                    d |= mid | (mid << 1) | (mid >> 1) | (lft >> 31) | (rgt << 31);
                }
            }
            nv[k] = ls[idx] | (lw[idx] & d);     // strong |= weak & dilate8
        }
        __syncthreads();                          // two-phase: exact reference
#pragma unroll
        for (int k = 0; k < 8; ++k) ls[tid + k * 1024] = nv[k];
        __syncthreads();
    }

#pragma unroll
    for (int k = 0; k < 8; ++k) {
        const int idx = tid + k * 1024;
#if defined(USE_ASYNC_LDS)
        async_l2g(&gS[idx], &ls[idx]);
#else
        gS[idx] = ls[idx];
#endif
    }
#if defined(USE_ASYNC_LDS)
    wait_asynccnt0();
#endif
}

// ---------------- K3: mismatch popcount + WMMA wave reduction -------------
__global__ __launch_bounds__(256) void count_kernel(
    const uint32_t* __restrict__ S, float* __restrict__ counter)
{
    // 128 blocks x 256 threads = 32768 threads; 262144 words / 32768 = 8 each.
    // No divergence before the WMMA (EXEC must be all ones).
    const int gtid = (int)blockIdx.x * 256 + (int)threadIdx.x;
    int cnt = 0;
#pragma unroll
    for (int k = 0; k < 8; ++k) {
        const int i = gtid + k * 32768;
        cnt += __popc(S[i] ^ S[i + HALF_PLANES]);
    }
    const float wsum = wave_reduce_wmma((float)cnt);
    if ((threadIdx.x & 31u) == 0u)
        atomicAdd(counter, wsum);   // exact: integer-valued, total < 2^24
}

// ---------------- K4: finalize BCE ----------------------------------------
__global__ void finalize_kernel(const float* __restrict__ counter,
                                float* __restrict__ out)
{
    if (threadIdx.x == 0) {
        const float cost = -logf(1e-38f);      // per-mismatch BCE term
        out[0] = counter[0] * cost / (float)NPIX;
    }
}

// ---------------------------------------------------------------------------
extern "C" void kernel_launch(void* const* d_in, const int* in_sizes, int n_in,
                              void* d_out, int out_size, void* d_ws, size_t ws_size,
                              hipStream_t stream) {
    (void)in_sizes; (void)n_in; (void)out_size; (void)ws_size;
    const float* pred   = (const float*)d_in[0];
    const float* labels = (const float*)d_in[1];
    uint32_t* S  = (uint32_t*)d_ws;           // strong planes: 64 * 8192 words
    uint32_t* Wk = S + ALL_PLANES;            // weak planes:   64 * 8192 words
    float* counter = (float*)(Wk + ALL_PLANES);
    float* out = (float*)d_out;

    edge_kernel<<<dim3(IMG_W / 256, IMG_H, IMG_B * 2), dim3(256), 0, stream>>>(
        pred, labels, S, Wk, counter);
    hyst_kernel<<<dim3(IMG_B * 2), dim3(1024), 0, stream>>>(S, Wk);
    count_kernel<<<dim3(128), dim3(256), 0, stream>>>(S, counter);
    finalize_kernel<<<dim3(1), dim3(32), 0, stream>>>(counter, out);
}
